// DRNLayer_50165218018101
// MI455X (gfx1250) — matrix operand order repaired and checked by reading, not compiled, not run on hardware
//
#include <hip/hip_runtime.h>
#include <hip/hip_bf16.h>
#include <stdint.h>

// Problem constants (from reference): B=256, NU=64, NL=64, QU=64, QL=64
#define DB  256
#define DNU 64
#define DNL 64
#define DQU 64
#define DQL 64

#define LOG2E 1.4426950408889634f

typedef __attribute__((ext_vector_type(16))) _Float16 v16h;
typedef __attribute__((ext_vector_type(4)))  _Float16 v4h;
typedef __attribute__((ext_vector_type(8)))  float    v8f;

union ABu {
    v16h h;
    uint4 q[2];
};

// ---------------------------------------------------------------------------
// Kernel 1: T[jk][l][m] = exp(-w[jk] * ((m-l)/64)^2)  stored f16 row-major.
// exp(x) computed as exp2 with log2e folded into w -> single v_exp_f32.
// Row-major [l][m] is exactly the WMMA B-matrix load layout (lane = column l,
// 16 consecutive m per 32B chunk).
// ---------------------------------------------------------------------------
__global__ __launch_bounds__(256) void drn_prep_T(const float* __restrict__ weight,
                                                  _Float16* __restrict__ Tf16) {
    const int jk = blockIdx.x;               // j*NL + k
    const float wl = weight[jk] * LOG2E;
    _Float16* tile = Tf16 + (size_t)jk * (DQU * DQL);
    const int t = threadIdx.x;
    const int base = t * 16;                 // 16 contiguous halves (32B) per thread
    #pragma unroll
    for (int e = 0; e < 16; ++e) {
        const int idx = base + e;
        const int l = idx >> 6;
        const int m = idx & 63;
        const float d = (float)(m - l) * (1.0f / 64.0f);
        tile[idx] = (_Float16)__builtin_amdgcn_exp2f(-wl * d * d);
    }
}

// ---------------------------------------------------------------------------
// Kernel 2: convert P (f32 [B][NL][QL]) -> f16 same layout (A-matrix source).
// ---------------------------------------------------------------------------
__global__ __launch_bounds__(256) void drn_prep_P(const float* __restrict__ P,
                                                  _Float16* __restrict__ Pf16) {
    const int idx = blockIdx.x * 256 + threadIdx.x;      // over (B*NL*QL)/4
    const float4 v = reinterpret_cast<const float4*>(P)[idx];
    v4h h;
    h[0] = (_Float16)v.x; h[1] = (_Float16)v.y;
    h[2] = (_Float16)v.z; h[3] = (_Float16)v.w;
    *reinterpret_cast<v4h*>(Pf16 + (size_t)idx * 4) = h;
}

// ---------------------------------------------------------------------------
// Main kernel: grid = (NU, B/64); block = 128 threads = 4 waves (wave32).
// k unrolled x2: each iteration stages TWO 8KB T tiles (16KB) via
// global_load_async_to_lds_b128 double-buffering, issues 16 WMMAs, then does
// the fused log-accumulation with PAIRED logs:
//   log2(clip(pw_k)) + log2(clip(pw_k1)) == log2(clip(pw_k)*clip(pw_k1))
// halving the dominant v_log_f32 stream.
// ---------------------------------------------------------------------------
__device__ __forceinline__ void stage_T2_async(const char* gtile, int ldsBufBase, int tid) {
    // 16384 bytes / 128 threads = 128B/thread = 8 x b128 per thread
    #pragma unroll
    for (int i = 0; i < 8; ++i) {
        const unsigned ldsOff = (unsigned)(ldsBufBase + i * 2048 + tid * 16);
        const void* gaddr = gtile + i * 2048 + tid * 16;
        asm volatile("global_load_async_to_lds_b128 %0, %1, off"
                     :
                     : "v"(ldsOff), "v"(gaddr)
                     : "memory");
    }
}

__global__ __launch_bounds__(128) void drn_main(const _Float16* __restrict__ Pf16,
                                                const _Float16* __restrict__ Tf16,
                                                const float* __restrict__ bias_abs,
                                                const float* __restrict__ bias_q,
                                                const float* __restrict__ lam_abs,
                                                const float* __restrict__ lam_q,
                                                float* __restrict__ out) {
    extern __shared__ char smem[];            // 2 x 16384B (two T tiles per buffer)

    const int j      = blockIdx.x;            // upper neuron
    const int bblk   = blockIdx.y;            // batch block of 64
    const int tid    = threadIdx.x;
    const int wave   = tid >> 5;              // 0..3
    const int lane   = tid & 31;
    const int halfId = lane >> 4;             // 0: lanes 0-15, 1: lanes 16-31
    const int l16    = lane & 15;

    const int rowBase = bblk * 64 + wave * 16;        // batch row of M-tile

    v8f acc[4];
    #pragma unroll
    for (int nt = 0; nt < 4; ++nt) acc[nt] = (v8f)0.0f;

    // A-matrix per-lane base: lane<16 holds K{0..7,16..23}, lane>=16 K{8..15,24..31}
    const _Float16* Prow0 = Pf16 + (size_t)(rowBase + l16) * (DNL * DQL);
    const int aoff = halfId ? 8 : 0;
    const char* Tj = reinterpret_cast<const char*>(Tf16) + (size_t)j * (DNL * DQU * DQL) * 2;

    // B-fragment LDS byte offset within a tile (row l is 128B)
    const int bRow = l16 * 128 + (halfId ? 16 : 0);

    stage_T2_async(Tj, 0, tid);               // prologue: tiles k=0,1 -> buf 0

    for (int kk = 0; kk < DNL / 2; ++kk) {
        const int buf = kk & 1;
        asm volatile("s_wait_asynccnt 0x0" ::: "memory");
        __syncthreads();                       // buffer[buf] ready for all waves
        if (kk + 1 < DNL / 2)
            stage_T2_async(Tj + (size_t)(kk + 1) * 16384, (buf ^ 1) * 16384, tid);

        // A fragments for k=2kk and k=2kk+1 (rows are 128B apart, contiguous)
        const _Float16* prow = Prow0 + (2 * kk) * DQL;
        ABu a0, a1, a2, a3;
        a0.q[0] = *reinterpret_cast<const uint4*>(prow + aoff);
        a0.q[1] = *reinterpret_cast<const uint4*>(prow + aoff + 16);
        a1.q[0] = *reinterpret_cast<const uint4*>(prow + 32 + aoff);
        a1.q[1] = *reinterpret_cast<const uint4*>(prow + 32 + aoff + 16);
        a2.q[0] = *reinterpret_cast<const uint4*>(prow + 64 + aoff);
        a2.q[1] = *reinterpret_cast<const uint4*>(prow + 64 + aoff + 16);
        a3.q[0] = *reinterpret_cast<const uint4*>(prow + 96 + aoff);
        a3.q[1] = *reinterpret_cast<const uint4*>(prow + 96 + aoff + 16);

        const char* tb0 = smem + buf * 16384;          // tile k
        const char* tb1 = tb0 + 8192;                  // tile k+1
        v8f c0[4], c1[4];
        #pragma unroll
        for (int nt = 0; nt < 4; ++nt) {
            // B-matrix: lane n = column l, lanes 0-15 hold K{0..7,16..23}, lanes 16-31 K{8..15,24..31}
            const uint4* q0 = reinterpret_cast<const uint4*>(tb0 + nt * 2048 + bRow);
            const uint4* q1 = reinterpret_cast<const uint4*>(tb1 + nt * 2048 + bRow);
            ABu b0, b1, b2, b3;
            b0.q[0] = q0[0]; b0.q[1] = q0[2];
            b1.q[0] = q0[4]; b1.q[1] = q0[6];
            b2.q[0] = q1[0]; b2.q[1] = q1[2];
            b3.q[0] = q1[4]; b3.q[1] = q1[6];
            v8f t0 = (v8f)0.0f, t1 = (v8f)0.0f;
            t0 = __builtin_amdgcn_wmma_f32_16x16x32_f16(false, a0.h, false, b0.h,
                                                        (short)0, t0, false, false);
            c0[nt] = __builtin_amdgcn_wmma_f32_16x16x32_f16(false, a1.h, false, b1.h,
                                                            (short)0, t0, false, false);
            t1 = __builtin_amdgcn_wmma_f32_16x16x32_f16(false, a2.h, false, b2.h,
                                                        (short)0, t1, false, false);
            c1[nt] = __builtin_amdgcn_wmma_f32_16x16x32_f16(false, a3.h, false, b3.h,
                                                            (short)0, t1, false, false);
        }

        // Pass 1: clip both tiles and pair-multiply (core VALU, independent ops)
        #pragma unroll
        for (int nt = 0; nt < 4; ++nt)
            #pragma unroll
            for (int r = 0; r < 8; ++r)
                c0[nt][r] = __builtin_amdgcn_fmed3f(c0[nt][r], 1e-15f, 1e15f) *
                            __builtin_amdgcn_fmed3f(c1[nt][r], 1e-15f, 1e15f);
        // Pass 2: 32 independent v_log_f32 (TRANS pipelines back-to-back)
        #pragma unroll
        for (int nt = 0; nt < 4; ++nt)
            #pragma unroll
            for (int r = 0; r < 8; ++r)
                acc[nt][r] += __builtin_amdgcn_logf(c0[nt][r]);   // log2

        __syncthreads();   // all waves done reading buffer[buf] before refill
    }

    // exponent_B[j, l] scaled by log2e: softmax base-2 of (log2e*x) == softmax(x)
    const float ba = bias_abs[j];
    const float bq = bias_q[j];
    const float la = lam_abs[j];
    const float lq = lam_q[j];
    float eb2[4];
    #pragma unroll
    for (int nt = 0; nt < 4; ++nt) {
        const float sq = (float)(nt * 16 + l16) * (1.0f / DQU);
        const float dq = sq - lq;
        const float da = sq - la;
        eb2[nt] = (-bq * dq * dq - ba * fabsf(da)) * LOG2E;
    }

    // softmax over l (64 columns = 4 nt-tiles x 16 lanes of this half-wave).
    // Row M=r (lanes 0-15) / M=8+r (lanes 16-31) per WMMA C/D layout; xor masks
    // 1,2,4,8 keep the reduction inside the owning 16-lane half.
    #pragma unroll
    for (int r = 0; r < 8; ++r) {
        float v0 = acc[0][r] + eb2[0];
        float v1 = acc[1][r] + eb2[1];
        float v2 = acc[2][r] + eb2[2];
        float v3 = acc[3][r] + eb2[3];
        float m = fmaxf(fmaxf(v0, v1), fmaxf(v2, v3));
        #pragma unroll
        for (int off = 1; off < 16; off <<= 1)
            m = fmaxf(m, __shfl_xor(m, off, 32));
        const float e0 = __builtin_amdgcn_exp2f(v0 - m);
        const float e1 = __builtin_amdgcn_exp2f(v1 - m);
        const float e2 = __builtin_amdgcn_exp2f(v2 - m);
        const float e3 = __builtin_amdgcn_exp2f(v3 - m);
        float s = (e0 + e1) + (e2 + e3);
        #pragma unroll
        for (int off = 1; off < 16; off <<= 1)
            s += __shfl_xor(s, off, 32);
        const float inv = __builtin_amdgcn_rcpf(s);
        const int i = rowBase + halfId * 8 + r;                 // batch index
        float* orow = out + ((size_t)i * DNU + j) * DQU + l16;  // out[i][j][l]
        orow[0]  = e0 * inv;
        orow[16] = e1 * inv;
        orow[32] = e2 * inv;
        orow[48] = e3 * inv;
    }
}

// ---------------------------------------------------------------------------
extern "C" void kernel_launch(void* const* d_in, const int* in_sizes, int n_in,
                              void* d_out, int out_size, void* d_ws, size_t ws_size,
                              hipStream_t stream) {
    (void)in_sizes; (void)n_in; (void)out_size; (void)ws_size;
    const float* P          = (const float*)d_in[0];
    const float* weight     = (const float*)d_in[1];
    const float* bias_abs   = (const float*)d_in[2];
    const float* bias_q     = (const float*)d_in[3];
    const float* lambda_abs = (const float*)d_in[4];
    const float* lambda_q   = (const float*)d_in[5];
    float* out = (float*)d_out;

    // Workspace layout: [ T f16 : 32MB ][ P f16 : 2MB ]  (L2 is 192MB -> resident)
    _Float16* Tf16 = reinterpret_cast<_Float16*>(d_ws);
    _Float16* Pf16 = reinterpret_cast<_Float16*>(
        reinterpret_cast<char*>(d_ws) + (size_t)DNU * DNL * DQU * DQL * sizeof(_Float16));

    drn_prep_T<<<DNU * DNL, 256, 0, stream>>>(weight, Tf16);
    drn_prep_P<<<(DB * DNL * DQL) / (256 * 4), 256, 0, stream>>>(P, Pf16);

    dim3 grid(DNU, DB / 64);
    drn_main<<<grid, 128, 2 * 16384, stream>>>(Pf16, Tf16, bias_abs, bias_q,
                                               lambda_abs, lambda_q, out);
}
